// ParallelTransformerBlock_72103910966035
// MI455X (gfx1250) — matrix operand order repaired and checked by reading, not compiled, not run on hardware
//
#include <hip/hip_runtime.h>
#include <stdint.h>

// ---------------------------------------------------------------------------
// Types for CDNA5 WMMA (wave32, 16x16x32 bf16 -> f32)
// ---------------------------------------------------------------------------
typedef __attribute__((ext_vector_type(16))) __bf16 v16bf;
typedef __attribute__((ext_vector_type(8)))  float  v8f;

union FragU { v16bf v; uint4 q[2]; };

// Model dims
#define N_SEQ     2048
#define DIM       2048
#define DIM_HEAD  128
#define HEADS     16
#define ATTN_INNER (DIM_HEAD * HEADS)      // 2048
#define FF_INNER  (DIM * 4)                // 8192
#define FUSED_OUT (ATTN_INNER + DIM_HEAD + DIM_HEAD + 2 * FF_INNER) // 18688

static __device__ __forceinline__ unsigned short f2bf(float f) {
    union { float f; unsigned int u; } x; x.f = f;
    unsigned int u = x.u;
    unsigned int r = (u + 0x7fffu + ((u >> 16) & 1u)) >> 16;
    return (unsigned short)r;
}
static __device__ __forceinline__ float bf2f(unsigned short h) {
    union { unsigned int u; float f; } x; x.u = ((unsigned int)h) << 16;
    return x.f;
}

// A-fragment loader (16-bit A 16x32, per ISA 7.12.2):
//   lanes 0-15  : row M=lane,    K = {0..7, 16..23}
//   lanes 16-31 : row M=lane-16, K = {8..15, 24..31}
static __device__ __forceinline__ v16bf load_frag_a(const unsigned short* base, int ld, int k0) {
    int lane = threadIdx.x & 31;
    int mk = lane & 15;
    int ko = (lane >> 4) * 8;
    const unsigned short* p = base + (size_t)mk * ld + k0 + ko;
    FragU u;
    u.q[0] = *reinterpret_cast<const uint4*>(p);        // K = ko..ko+7
    u.q[1] = *reinterpret_cast<const uint4*>(p + 16);   // K = ko+16..ko+23
    return u.v;
}

// B-fragment loader (16-bit B 32x16): lanes 0-15 hold col N=lane, K={0..7,16..23};
// lanes 16-31 hold col N=lane-16, K={8..15,24..31} (same K order as A).
// base: "column-major" storage, row index = N, contiguous K. ld = stride per N.
static __device__ __forceinline__ v16bf load_frag_b(const unsigned short* base, int ld, int k0) {
    int lane = threadIdx.x & 31;
    int mk = lane & 15;
    int ko = (lane >> 4) * 8;
    const unsigned short* p = base + (size_t)mk * ld + k0 + ko;
    FragU u;
    u.q[0] = *reinterpret_cast<const uint4*>(p);
    u.q[1] = *reinterpret_cast<const uint4*>(p + 16);
    return u.v;
}

static __device__ __forceinline__ v8f wmma_bf16(v16bf a, v16bf b, v8f c) {
    return __builtin_amdgcn_wmma_f32_16x16x32_bf16(false, a, false, b, (short)0, c, false, false);
}

// ---------------------------------------------------------------------------
// K1: f32 -> bf16 conversion of x
// ---------------------------------------------------------------------------
__global__ __launch_bounds__(256) void convert_bf16_kernel(const float* __restrict__ src,
                                                           unsigned short* __restrict__ dst,
                                                           int n) {
    int i = (blockIdx.x * 256 + threadIdx.x) * 4;
    if (i + 3 < n) {
        float4 v = *reinterpret_cast<const float4*>(src + i);
        ushort4 o;
        o.x = f2bf(v.x); o.y = f2bf(v.y); o.z = f2bf(v.z); o.w = f2bf(v.w);
        *reinterpret_cast<ushort4*>(dst + i) = o;
    }
}

// ---------------------------------------------------------------------------
// Stage one 32(K) x 128(N) f32 panel of B into LDS as bf16, K-major.
// Thread t: 4 consecutive cols (nq), 4 consecutive k rows (kq); float4 loads
// fully coalesced across lanes; b64 LDS stores.
// ---------------------------------------------------------------------------
static __device__ __forceinline__ void stage_panel(const float* __restrict__ B, int N, int bn,
                                                   int k0, unsigned short* Bt, int tid) {
    const int nq = (tid & 31) * 4;
    const int kq = (tid >> 5) * 4;
    const float* bp = B + (size_t)(k0 + kq) * N + bn + nq;
    unsigned short h[4][4];
#pragma unroll
    for (int j = 0; j < 4; j++) {
        float4 v = *reinterpret_cast<const float4*>(bp + (size_t)j * N);
        h[0][j] = f2bf(v.x); h[1][j] = f2bf(v.y); h[2][j] = f2bf(v.z); h[3][j] = f2bf(v.w);
    }
#pragma unroll
    for (int c = 0; c < 4; c++)
        *reinterpret_cast<uint2*>(&Bt[(nq + c) * 40 + kq]) =
            *reinterpret_cast<const uint2*>(&h[c][0]);
}

// ---------------------------------------------------------------------------
// Tiled bf16 WMMA GEMM:  C(MxN) = A(MxK, bf16 row-major) * B(KxN, f32 row-major)
// Double-buffered LDS B panels: stage panel i+1 while WMMAs consume panel i.
// Workgroup: 256 threads (8 waves), 128x128 C tile, 32-deep K panels.
// Wave (wm,wn) grid 4x2: each wave -> 32 rows x 64 cols = 2x4 WMMA tiles.
// MODE: 0 = bf16 store, 1 = f32 store, 2 = f32 accumulate-add
// ---------------------------------------------------------------------------
template <int MODE>
__global__ __launch_bounds__(256) void gemm_bf16_kernel(const unsigned short* __restrict__ A,
                                                        const float* __restrict__ B,
                                                        void* __restrict__ out,
                                                        int M, int N, int K) {
    __shared__ unsigned short Bt[2][128 * 40];   // ping-pong, [n][k] K-major, padded

    const int tid  = threadIdx.x;
    const int w    = tid >> 5;
    const int lane = tid & 31;
    const int bm = blockIdx.y * 128;
    const int bn = blockIdx.x * 128;
    const int wm = (w & 3) * 32;     // wave row offset
    const int wn = (w >> 2) * 64;    // wave col offset

    v8f acc[2][4];
#pragma unroll
    for (int i = 0; i < 2; i++)
#pragma unroll
        for (int j = 0; j < 4; j++) acc[i][j] = (v8f)(0.0f);

    stage_panel(B, N, bn, 0, Bt[0], tid);
    __syncthreads();

    int buf = 0;
    for (int k0 = 0; k0 < K; k0 += 32, buf ^= 1) {
        // ---- stage next panel into the other buffer (overlaps the WMMAs) ----
        if (k0 + 32 < K) {
            stage_panel(B, N, bn, k0 + 32, Bt[buf ^ 1], tid);
            if (k0 + 64 < K)
                __builtin_prefetch(B + (size_t)(k0 + 64) * N + bn + (tid & 31) * 4, 0, 1);
        }

        // ---- fragments + 8 WMMAs from current buffer ----
        v16bf af[2];
#pragma unroll
        for (int mt = 0; mt < 2; mt++)
            af[mt] = load_frag_a(A + (size_t)(bm + wm + mt * 16) * K, K, k0);
        v16bf bf[4];
#pragma unroll
        for (int nt = 0; nt < 4; nt++)
            bf[nt] = load_frag_b(&Bt[buf][(wn + nt * 16) * 40], 40, 0);
#pragma unroll
        for (int mt = 0; mt < 2; mt++)
#pragma unroll
            for (int nt = 0; nt < 4; nt++)
                acc[mt][nt] = wmma_bf16(af[mt], bf[nt], acc[mt][nt]);

        __syncthreads();   // next-panel stores visible; current-buffer reads done
    }

    // ---- epilogue (C layout: lanes 0-15 -> M=r, lanes 16-31 -> M=r+8, N=lane&15) ----
    const int mk   = lane & 15;
    const int half = lane >> 4;
#pragma unroll
    for (int mt = 0; mt < 2; mt++)
#pragma unroll
        for (int nt = 0; nt < 4; nt++)
#pragma unroll
            for (int r = 0; r < 8; r++) {
                int row = bm + wm + mt * 16 + r + half * 8;
                int col = bn + wn + nt * 16 + mk;
                float v = acc[mt][nt][r];
                size_t idx = (size_t)row * N + col;
                if (MODE == 0) ((unsigned short*)out)[idx] = f2bf(v);
                else if (MODE == 1) ((float*)out)[idx] = v;
                else ((float*)out)[idx] += v;
            }
}

// ---------------------------------------------------------------------------
// K3: epilogue on fused GEMM output: rotary(q)*scale, rotary(k), v^T, swish-gate FF
// fused layout per row: [0,2048) q | [2048,2176) k | [2176,2304) v | [2304,18688) ff
// ---------------------------------------------------------------------------
__global__ __launch_bounds__(256) void fused_epilogue_kernel(const unsigned short* __restrict__ fused,
                                                             unsigned short* __restrict__ q_rot,
                                                             unsigned short* __restrict__ k_rot,
                                                             unsigned short* __restrict__ vT,
                                                             unsigned short* __restrict__ ffact) {
    const int COLS = ATTN_INNER + DIM_HEAD + DIM_HEAD + FF_INNER; // 10496 outputs/row
    int idx = blockIdx.x * 256 + threadIdx.x;
    int total = N_SEQ * COLS;
    if (idx >= total) return;
    int row = idx / COLS;
    int c   = idx - row * COLS;
    const unsigned short* frow = fused + (size_t)row * FUSED_OUT;

    if (c < ATTN_INNER) {                       // q with rotary + scale
        int d = c & 127, p = d & 63;
        float freq = (float)row * __expf(-(float)p * (9.210340371976184f / 64.0f));
        float sn, cs; __sincosf(freq, &sn, &cs);
        float t  = bf2f(frow[c]);
        float tp = bf2f(frow[(d < 64) ? c + 64 : c - 64]);
        float sign = (d < 64) ? -1.0f : 1.0f;
        float o = (t * cs + sign * tp * sn) * 0.08838834764831845f; // 128^-0.5
        q_rot[(size_t)row * ATTN_INNER + c] = f2bf(o);
    } else if (c < ATTN_INNER + DIM_HEAD) {     // k with rotary
        int d = c - ATTN_INNER, p = d & 63;
        float freq = (float)row * __expf(-(float)p * (9.210340371976184f / 64.0f));
        float sn, cs; __sincosf(freq, &sn, &cs);
        float t  = bf2f(frow[c]);
        float tp = bf2f(frow[c + ((d < 64) ? 64 : -64)]);
        float sign = (d < 64) ? -1.0f : 1.0f;
        k_rot[(size_t)row * DIM_HEAD + d] = f2bf(t * cs + sign * tp * sn);
    } else if (c < ATTN_INNER + 2 * DIM_HEAD) { // v, store transposed [dim][seq]
        int d = c - (ATTN_INNER + DIM_HEAD);
        vT[(size_t)d * N_SEQ + row] = frow[c];
    } else {                                    // ff: x * swish(gate)
        int j = c - (ATTN_INNER + 2 * DIM_HEAD);
        float xv = bf2f(frow[ATTN_INNER + 2 * DIM_HEAD + j]);
        float g  = bf2f(frow[ATTN_INNER + 2 * DIM_HEAD + FF_INNER + j]);
        float sw = g / (1.0f + __expf(-g));
        ffact[(size_t)row * FF_INNER + j] = f2bf(xv * sw);
    }
}

// ---------------------------------------------------------------------------
// K4: causal flash attention (multi-query), bf16 WMMA, barrier-free.
// grid = HEADS * 32 workgroups; 128 threads (4 waves); wave owns 16 q rows.
// P tile is wave-private LDS; same-wave LDS ops are in order (DScnt), so no
// __syncthreads needed -> each wave runs its own (shorter) causal trip count.
// ---------------------------------------------------------------------------
__global__ __launch_bounds__(128) void flash_attn_kernel(const unsigned short* __restrict__ q_rot,
                                                         const unsigned short* __restrict__ k_rot,
                                                         const unsigned short* __restrict__ vT,
                                                         unsigned short* __restrict__ attn_o) {
    __shared__ unsigned short P[4][16 * 40];   // per-wave P tile, [row][key], padded

    const int head = blockIdx.x >> 5;          // /32
    const int rb   = blockIdx.x & 31;
    const int wave = threadIdx.x >> 5;
    const int lane = threadIdx.x & 31;
    const int mk   = lane & 15;
    const int half = lane >> 4;
    const int rbase = rb * 64 + wave * 16;

    // Q fragments (16 rows x 128 dims), resident in registers
    v16bf qf[4];
#pragma unroll
    for (int kf = 0; kf < 4; kf++)
        qf[kf] = load_frag_a(q_rot + (size_t)rbase * ATTN_INNER + head * DIM_HEAD, ATTN_INNER, kf * 32);

    v8f o[8];
#pragma unroll
    for (int nt = 0; nt < 8; nt++) o[nt] = (v8f)(0.0f);
    float mi[8], li[8];
#pragma unroll
    for (int r = 0; r < 8; r++) { mi[r] = -3.0e38f; li[r] = 0.0f; }

    const int nblocks = (rbase + 15) / 32 + 1;   // per-wave causal trip count
    for (int jb = 0; jb < nblocks; jb++) {
        const int kb = jb * 32;

        // ---- S = Q K^T for 32 keys (two 16x16 tiles) ----
        v8f S0 = (v8f)(0.0f), S1 = (v8f)(0.0f);
#pragma unroll
        for (int kf = 0; kf < 4; kf++) {
            v16bf b0 = load_frag_b(k_rot + (size_t)kb * DIM_HEAD,        DIM_HEAD, kf * 32);
            v16bf b1 = load_frag_b(k_rot + (size_t)(kb + 16) * DIM_HEAD, DIM_HEAD, kf * 32);
            S0 = wmma_bf16(qf[kf], b0, S0);
            S1 = wmma_bf16(qf[kf], b1, S1);
        }

        // ---- causal mask + online softmax ----
        const int c0 = kb + mk, c1 = c0 + 16;
#pragma unroll
        for (int r = 0; r < 8; r++) {
            int row = rbase + r + half * 8;
            float s0 = (c0 > row) ? -1.0e10f : S0[r];
            float s1 = (c1 > row) ? -1.0e10f : S1[r];
            float mr = fmaxf(s0, s1);
#pragma unroll
            for (int d = 1; d < 16; d <<= 1) mr = fmaxf(mr, __shfl_xor(mr, d, 32));
            float mnew = fmaxf(mi[r], mr);
            float sc = __expf(mi[r] - mnew);
            float p0 = __expf(s0 - mnew);
            float p1 = __expf(s1 - mnew);
            float rs = p0 + p1;
#pragma unroll
            for (int d = 1; d < 16; d <<= 1) rs += __shfl_xor(rs, d, 32);
            li[r] = li[r] * sc + rs;
            mi[r] = mnew;
#pragma unroll
            for (int nt = 0; nt < 8; nt++) o[nt][r] *= sc;
            int rr = r + half * 8;
            P[wave][rr * 40 + mk]      = f2bf(p0);
            P[wave][rr * 40 + 16 + mk] = f2bf(p1);
        }

        // ---- O += P V  (P: 16x32 A-frag from wave-private LDS; in-order DS) ----
        v16bf pf = load_frag_a(&P[wave][0], 40, 0);
#pragma unroll
        for (int nt = 0; nt < 8; nt++) {
            v16bf vf = load_frag_b(vT + (size_t)(nt * 16) * N_SEQ, N_SEQ, kb);
            o[nt] = wmma_bf16(pf, vf, o[nt]);
        }
    }

    // ---- finalize: O / l, write bf16 [seq][head*128+dim] ----
#pragma unroll
    for (int nt = 0; nt < 8; nt++)
#pragma unroll
        for (int r = 0; r < 8; r++) {
            int row = rbase + r + half * 8;
            int col = head * DIM_HEAD + nt * 16 + mk;
            attn_o[(size_t)row * ATTN_INNER + col] = f2bf(o[nt][r] / li[r]);
        }
}

// ---------------------------------------------------------------------------
// Launch
// ---------------------------------------------------------------------------
extern "C" void kernel_launch(void* const* d_in, const int* in_sizes, int n_in,
                              void* d_out, int out_size, void* d_ws, size_t ws_size,
                              hipStream_t stream) {
    const float* x          = (const float*)d_in[0];
    const float* w_fused    = (const float*)d_in[1];
    const float* w_attn_out = (const float*)d_in[2];
    const float* w_ff_out   = (const float*)d_in[3];
    float* out = (float*)d_out;

    // workspace layout (bf16 buffers)
    char* ws = (char*)d_ws;
    size_t off = 0;
    unsigned short* xb     = (unsigned short*)(ws + off); off += (size_t)N_SEQ * DIM * 2;
    unsigned short* fused  = (unsigned short*)(ws + off); off += (size_t)N_SEQ * FUSED_OUT * 2;
    unsigned short* q_rot  = (unsigned short*)(ws + off); off += (size_t)N_SEQ * ATTN_INNER * 2;
    unsigned short* k_rot  = (unsigned short*)(ws + off); off += (size_t)N_SEQ * DIM_HEAD * 2;
    unsigned short* vT     = (unsigned short*)(ws + off); off += (size_t)DIM_HEAD * N_SEQ * 2;
    unsigned short* ffact  = (unsigned short*)(ws + off); off += (size_t)N_SEQ * FF_INNER * 2;
    unsigned short* attn_o = (unsigned short*)(ws + off); off += (size_t)N_SEQ * ATTN_INNER * 2;
    (void)ws_size; (void)n_in; (void)in_sizes; (void)out_size;

    // 1) x -> bf16
    {
        int n = N_SEQ * DIM;
        convert_bf16_kernel<<<(n / 4 + 255) / 256, 256, 0, stream>>>(x, xb, n);
    }
    // 2) fused = x @ w_fused   (2048 x 18688, K=2048) -> bf16
    gemm_bf16_kernel<0><<<dim3(FUSED_OUT / 128, N_SEQ / 128), 256, 0, stream>>>(
        xb, w_fused, (void*)fused, N_SEQ, FUSED_OUT, DIM);
    // 3) rotary / v^T / swish-gate epilogue
    {
        int total = N_SEQ * (ATTN_INNER + 2 * DIM_HEAD + FF_INNER);
        fused_epilogue_kernel<<<(total + 255) / 256, 256, 0, stream>>>(fused, q_rot, k_rot, vT, ffact);
    }
    // 4) causal flash attention
    flash_attn_kernel<<<HEADS * 32, 128, 0, stream>>>(q_rot, k_rot, vT, attn_o);
    // 5) out = attn_o @ w_attn_out (f32 store)
    gemm_bf16_kernel<1><<<dim3(DIM / 128, N_SEQ / 128), 256, 0, stream>>>(
        attn_o, w_attn_out, (void*)out, N_SEQ, DIM, ATTN_INNER);
    // 6) out += ffact @ w_ff_out (f32 accumulate)
    gemm_bf16_kernel<2><<<dim3(DIM / 128, N_SEQ / 128), 256, 0, stream>>>(
        ffact, w_ff_out, (void*)out, N_SEQ, DIM, FF_INNER);
}
